// MultiBoxLoss2_73778948210753
// MI455X (gfx1250) — compile-verified, matrix operand order checked by reading, not run on hardware
//
#include <hip/hip_runtime.h>

#define NUM_CLASSES 81
#define B_    64
#define P_    24564
#define NOBJ_ 32

#define LOSS_BLOCK 128   // rows (priors) per loss block == threads per loss block

typedef __attribute__((ext_vector_type(16))) _Float16 v16h;
typedef __attribute__((ext_vector_type(8)))  float    v8f;

// ---------------------------------------------------------------------------
// Async global->LDS staging (CDNA5 GLOBAL_LOAD_ASYNC_TO_LDS_B128, ASYNCcnt).
// The builtin takes (v4i AS1* src, v4i AS3* dst, imm offset, imm cpol), where
// v4i is the gcc-style vector: __attribute__((vector_size(16))) int.
// Generic LDS pointer low 32 bits == LDS byte offset (ISA §10.2).
// ---------------------------------------------------------------------------
#if __has_builtin(__builtin_amdgcn_global_load_async_to_lds_b128) && \
    __has_builtin(__builtin_amdgcn_s_wait_asynccnt)
#define HAS_ASYNC_LDS 1
typedef int v4i_ __attribute__((vector_size(16)));
typedef __attribute__((address_space(1))) v4i_ glb_v4i;
typedef __attribute__((address_space(3))) v4i_ lds_v4i;
__device__ __forceinline__ glb_v4i* to_glb(const void* p) {
  return (glb_v4i*)(unsigned long long)p;
}
__device__ __forceinline__ lds_v4i* to_lds(void* p) {
  return (lds_v4i*)(unsigned int)(unsigned long long)p;
}
#else
#define HAS_ASYNC_LDS 0
#endif

// ---------------------------------------------------------------------------
// Block-wide (128 threads) sum of f32 partials via V_WMMA_F32_16X16X32_F16.
// Wave 0 packs 128 values, 4 f16 K-slots per lane: lane l<16 -> row l, K0..3;
// lane l+16 -> row l, K8..11 (distinct (M,K) slots per the CDNA5 A layout).
// B = ones 32x16 -> D[m][n] = rowsum_m. Lane0 C regs hold rows 0..7, lane16
// rows 8..15; total = s(lane0) + s(lane16). Valid on tid==0. EXEC all-ones
// inside the tid<32 branch (whole wave 0 of a full block).
// ---------------------------------------------------------------------------
__device__ __forceinline__ float block_sum_wmma(float v, float* red, int tid) {
  red[tid] = v;
  __syncthreads();
  float total = 0.0f;
  if (tid < 32) {
    v16h a, ones;
#pragma unroll
    for (int i = 0; i < 16; ++i) { a[i] = (_Float16)0.0f; ones[i] = (_Float16)1.0f; }
#pragma unroll
    for (int i = 0; i < 4; ++i) a[i] = (_Float16)red[tid * 4 + i];
    v8f c = {};
    // 8 args: (neg_a, A, neg_b, B, c_mod, C, reuse_a, reuse_b)
    c = __builtin_amdgcn_wmma_f32_16x16x32_f16(false, a, false, ones,
                                               (short)0, c, false, false);
    float s = 0.0f;
#pragma unroll
    for (int i = 0; i < 8; ++i) s += c[i];
    float s16 = __shfl(s, 16, 32);   // lane16 holds rows 8..15
    total = s + s16;                 // on lane 0: all 16 row sums
  }
  __syncthreads();
  return total;
}

// ---------------------------------------------------------------------------
__global__ void zero_ws_kernel(unsigned int* p, int n) {
  int i = blockIdx.x * blockDim.x + threadIdx.x;
  if (i < n) p[i] = 0u;
}

// ---------------------------------------------------------------------------
// Per (b,p): IoU against all 32 truths; per-prior best (first-max tie-break via
// strict >); per-truth argmax over priors via packed u64 keys:
//   key = (iou_bits << 32) | (0xFFFFFFFF - p)   (iou >= 0 -> bits monotonic,
//   smaller p wins ties == jnp.argmax first-index semantics).
// LDS ds_max_u64 pre-reduction, then one global atomicMax per truth per block.
// ---------------------------------------------------------------------------
__global__ __launch_bounds__(256) void match_kernel(
    const float* __restrict__ priors, const float* __restrict__ targets,
    float* __restrict__ ov, int* __restrict__ bidx,
    unsigned long long* __restrict__ pkey) {
  __shared__ float tr[NOBJ_ * 5];
  __shared__ unsigned long long skey[NOBJ_];
  const int tid = threadIdx.x;
  const int b = blockIdx.y;
  const int p = blockIdx.x * 256 + tid;
  if (tid < NOBJ_ * 5) tr[tid] = targets[b * NOBJ_ * 5 + tid];
  if (tid < NOBJ_) skey[tid] = 0ULL;
  __syncthreads();
  if (p < P_) {
    float4 pr = reinterpret_cast<const float4*>(priors)[p];
    const float px1 = pr.x - pr.z * 0.5f, py1 = pr.y - pr.w * 0.5f;
    const float px2 = pr.x + pr.z * 0.5f, py2 = pr.y + pr.w * 0.5f;
    const float parea = pr.z * pr.w;
    float bestv = -1.0f; int besti = 0;
    const unsigned lowp = 0xFFFFFFFFu - (unsigned)p;
#pragma unroll 4
    for (int t = 0; t < NOBJ_; ++t) {
      const float tx1 = tr[t * 5 + 0], ty1 = tr[t * 5 + 1];
      const float tx2 = tr[t * 5 + 2], ty2 = tr[t * 5 + 3];
      float iw = fmaxf(fminf(tx2, px2) - fmaxf(tx1, px1), 0.0f);
      float ih = fmaxf(fminf(ty2, py2) - fmaxf(ty1, py1), 0.0f);
      const float inter = iw * ih;
      const float tarea = (tx2 - tx1) * (ty2 - ty1);
      const float iou = inter / (tarea + parea - inter);
      if (iou > bestv) { bestv = iou; besti = t; }
      const unsigned long long key =
          ((unsigned long long)__float_as_uint(iou) << 32) | (unsigned long long)lowp;
      atomicMax(&skey[t], key);
    }
    ov[(size_t)b * P_ + p] = bestv;
    bidx[(size_t)b * P_ + p] = besti;
  }
  __syncthreads();
  if (tid < NOBJ_) atomicMax(&pkey[b * NOBJ_ + tid], skey[tid]);
}

// Force best prior per truth: sequential over t per batch => last t wins ties.
__global__ void override_kernel(const unsigned long long* __restrict__ pkey,
                                float* __restrict__ ov, int* __restrict__ bidx) {
  const int b = blockIdx.x * blockDim.x + threadIdx.x;
  if (b >= B_) return;
  for (int t = 0; t < NOBJ_; ++t) {
    const unsigned long long k = pkey[b * NOBJ_ + t];
    const unsigned p = 0xFFFFFFFFu - (unsigned)(k & 0xFFFFFFFFULL);
    if (p < (unsigned)P_) {
      ov[(size_t)b * P_ + p] = 2.0f;
      bidx[(size_t)b * P_ + p] = t;
    }
  }
}

// ---------------------------------------------------------------------------
// Fused loss. The 509 MB conf_data stream is the roofline term: each block
// asynchronously stages its contiguous 128-row x 81-float tile (41.5 KB) into
// LDS with coalesced b128 async copies (overlapped with the encode/smooth-L1
// work), then computes the stable log-softmax CE from LDS (81 = odd stride ->
// bank-conflict-free). Block partial sums reduced via WMMA.
// ---------------------------------------------------------------------------
__global__ __launch_bounds__(LOSS_BLOCK) void loss_kernel(
    const float* __restrict__ loc_data, const float* __restrict__ conf_data,
    const float* __restrict__ priors, const float* __restrict__ targets,
    const float* __restrict__ ov, const int* __restrict__ bidx,
    float* __restrict__ ce_out, int* __restrict__ numpos,
    float* __restrict__ gsum) {
  __shared__ float tile[LOSS_BLOCK * NUM_CLASSES];   // 41,472 B
  __shared__ float tr[NOBJ_ * 5];
  __shared__ float red[LOSS_BLOCK];
  __shared__ int cnt;
  const int tid = threadIdx.x;
  const int b = blockIdx.y;
  const int p0 = blockIdx.x * LOSS_BLOCK;
  const int p = p0 + tid;
  const int rows = min(LOSS_BLOCK, P_ - p0);        // 128 or 116 (both %4==0)
  const int chunks = (rows * NUM_CLASSES) >> 2;     // whole float4s, 16B aligned

  // --- stage conf tile: coalesced, async if available -----------------------
  const float* gtile = conf_data + ((size_t)b * P_ + p0) * NUM_CLASSES;
#if HAS_ASYNC_LDS
  for (int j = tid; j < chunks; j += LOSS_BLOCK) {
    __builtin_amdgcn_global_load_async_to_lds_b128(
        to_glb((const void*)(gtile + 4 * (size_t)j)),
        to_lds((void*)(tile + 4 * j)), 0, 0);
  }
#else
  for (int j = tid; j < chunks; j += LOSS_BLOCK) {
    reinterpret_cast<float4*>(tile)[j] =
        reinterpret_cast<const float4*>(gtile)[j];
  }
#endif

  if (tid < NOBJ_ * 5) tr[tid] = targets[b * NOBJ_ * 5 + tid];
  if (tid == 0) cnt = 0;
  __syncthreads();   // tr[] visible; async copy still in flight

  // --- matching/encode/smooth-L1 (independent of conf tile) ----------------
  float ll = 0.0f, pl = 0.0f;
  int conf_t = 0;
  bool pos = false;
  if (p < P_) {
    const size_t bp = (size_t)b * P_ + p;
    const float bo = ov[bp];
    const int t = bidx[bp];
    conf_t = (bo < 0.5f) ? 0 : ((int)tr[t * 5 + 4] + 1);
    pos = conf_t > 0;
    if (pos) {
      float4 pr = reinterpret_cast<const float4*>(priors)[p];
      const float mnx = tr[t * 5 + 0], mny = tr[t * 5 + 1];
      const float mxx = tr[t * 5 + 2], mxy = tr[t * 5 + 3];
      const float gx = ((mnx + mxx) * 0.5f - pr.x) / (0.1f * pr.z);
      const float gy = ((mny + mxy) * 0.5f - pr.y) / (0.1f * pr.w);
      const float gw = __logf((mxx - mnx) / pr.z) * 5.0f;   // /0.2
      const float gh = __logf((mxy - mny) / pr.w) * 5.0f;
      float4 ld = reinterpret_cast<const float4*>(loc_data)[bp];
      const float d0 = fabsf(ld.x - gx), d1 = fabsf(ld.y - gy);
      const float d2 = fabsf(ld.z - gw), d3 = fabsf(ld.w - gh);
      ll  = (d0 < 1.0f) ? 0.5f * d0 * d0 : d0 - 0.5f;
      ll += (d1 < 1.0f) ? 0.5f * d1 * d1 : d1 - 0.5f;
      ll += (d2 < 1.0f) ? 0.5f * d2 * d2 : d2 - 0.5f;
      ll += (d3 < 1.0f) ? 0.5f * d3 * d3 : d3 - 0.5f;
      atomicAdd(&cnt, 1);
    }
  }

  // --- wait for tile, then CE from LDS -------------------------------------
#if HAS_ASYNC_LDS
  __builtin_amdgcn_s_wait_asynccnt(0);
#endif
  __syncthreads();

  if (p < P_) {
    const size_t bp = (size_t)b * P_ + p;
    const float* row = tile + tid * NUM_CLASSES;
    float mx = -3.0e38f;
    for (int c = 0; c < NUM_CLASSES; ++c) mx = fmaxf(mx, row[c]);
    float se = 0.0f;
    for (int c = 0; c < NUM_CLASSES; ++c) se += __expf(row[c] - mx);
    const float xt = row[conf_t];
    float ce = fmaxf(__logf(se) + mx - xt, 0.0f);  // >= 0 (radix-select invariant)
    pl = pos ? ce : 0.0f;
    ce_out[bp] = pos ? 0.0f : ce;
  }
  __syncthreads();
  if (tid == 0 && cnt) atomicAdd(&numpos[b], cnt);

  const float lls = block_sum_wmma(ll, red, tid);
  if (tid == 0 && lls != 0.0f) atomicAdd(&gsum[0], lls);
  const float pls = block_sum_wmma(pl, red, tid);
  if (tid == 0 && pls != 0.0f) atomicAdd(&gsum[1], pls);
}

// ---------------------------------------------------------------------------
// Hard-negative mining: per row, k = min(3*num_pos, P-1); kth-largest of the
// nonnegative CE row via 31-pass MSB-down radix select (row is L2-resident),
// then exact tie-aware top-k sum: sum(e>kth) + (k - cnt_gt)*kth.
// ---------------------------------------------------------------------------
__global__ __launch_bounds__(256) void negmine_kernel(
    const float* __restrict__ ce, const int* __restrict__ numpos,
    float* __restrict__ gsum) {
  __shared__ unsigned prefix;
  __shared__ int remaining;
  __shared__ int cnt;
  __shared__ float fsum;
  const int tid = threadIdx.x;
  const int b = blockIdx.x;
  const float* row = ce + (size_t)b * P_;
  int k = numpos[b] * 3;
  if (k > P_ - 1) k = P_ - 1;
  if (k <= 0) return;  // uniform per block
  if (tid == 0) { prefix = 0u; remaining = k; }
  __syncthreads();
  for (int bit = 30; bit >= 0; --bit) {
    if (tid == 0) cnt = 0;
    __syncthreads();
    const unsigned cand = prefix | (1u << bit);
    const unsigned himask = ~((1u << bit) - 1u);
    int c = 0;
    for (int i = tid; i < P_; i += 256) {
      const unsigned e = __float_as_uint(row[i]);
      c += ((e & himask) == cand) ? 1 : 0;
    }
    if (c) atomicAdd(&cnt, c);
    __syncthreads();
    if (tid == 0) {
      if (cnt >= remaining) prefix = cand;
      else remaining -= cnt;
    }
    __syncthreads();
  }
  const unsigned kb = prefix;
  const float kf = __uint_as_float(kb);
  if (tid == 0) { cnt = 0; fsum = 0.0f; }
  __syncthreads();
  float s = 0.0f; int c2 = 0;
  for (int i = tid; i < P_; i += 256) {
    const float v = row[i];
    if (__float_as_uint(v) > kb) { s += v; ++c2; }
  }
  atomicAdd(&fsum, s);
  if (c2) atomicAdd(&cnt, c2);
  __syncthreads();
  if (tid == 0) atomicAdd(&gsum[2], fsum + (float)(k - cnt) * kf);
}

__global__ void finalize_kernel(const int* __restrict__ numpos,
                                const float* __restrict__ gsum,
                                float* __restrict__ out) {
  if (blockIdx.x == 0 && threadIdx.x == 0) {
    float N = 0.0f;
    for (int b = 0; b < B_; ++b) N += (float)numpos[b];
    out[0] = gsum[0] / N;
    out[1] = (gsum[1] + gsum[2]) / N;
  }
}

// ---------------------------------------------------------------------------
extern "C" void kernel_launch(void* const* d_in, const int* in_sizes, int n_in,
                              void* d_out, int out_size, void* d_ws, size_t ws_size,
                              hipStream_t stream) {
  (void)in_sizes; (void)n_in; (void)out_size; (void)ws_size;
  const float* loc_data  = (const float*)d_in[0];
  const float* conf_data = (const float*)d_in[1];
  const float* priors    = (const float*)d_in[2];
  const float* targets   = (const float*)d_in[3];

  char* ws = (char*)d_ws;
  size_t off = 0;
  float* ov = (float*)(ws + off);              off += (size_t)B_ * P_ * sizeof(float);
  int* bidx = (int*)(ws + off);                off += (size_t)B_ * P_ * sizeof(int);
  float* ce = (float*)(ws + off);              off += (size_t)B_ * P_ * sizeof(float);
  unsigned long long* pkey =
      (unsigned long long*)(ws + off);         off += (size_t)B_ * NOBJ_ * sizeof(unsigned long long);
  int* numpos = (int*)(ws + off);              off += (size_t)B_ * sizeof(int);
  float* gsum = (float*)(ws + off);            off += 4 * sizeof(float);

  // Zero the accumulator region (pkey | numpos | gsum are contiguous).
  const int zero_ints = B_ * NOBJ_ * 2 + B_ + 4;
  zero_ws_kernel<<<(zero_ints + 255) / 256, 256, 0, stream>>>((unsigned*)pkey, zero_ints);

  dim3 mgrid((P_ + 255) / 256, B_);
  match_kernel<<<mgrid, 256, 0, stream>>>(priors, targets, ov, bidx, pkey);
  override_kernel<<<1, 64, 0, stream>>>(pkey, ov, bidx);
  dim3 lgrid((P_ + LOSS_BLOCK - 1) / LOSS_BLOCK, B_);
  loss_kernel<<<lgrid, LOSS_BLOCK, 0, stream>>>(loc_data, conf_data, priors, targets,
                                                ov, bidx, ce, numpos, gsum);
  negmine_kernel<<<B_, 256, 0, stream>>>(ce, numpos, gsum);
  finalize_kernel<<<1, 64, 0, stream>>>(numpos, gsum, (float*)d_out);
}
